// TwoStreamAttention_28595892257531
// MI455X (gfx1250) — matrix operand order repaired and checked by reading, not compiled, not run on hardware
//
#include <hip/hip_runtime.h>
#include <hip/hip_bf16.h>

// ---------------------------------------------------------------------------
// Problem dims (compile-time constants from the reference)
// ---------------------------------------------------------------------------
#define S_LEN   2048
#define BATCH_N 2
#define NHEAD   16
#define DHEAD   64
#define DMODEL  1024
#define MROWS   (S_LEN * BATCH_N)   // 4096 flattened (i,b) rows

typedef __attribute__((ext_vector_type(16))) __bf16 v16bf;
typedef __attribute__((ext_vector_type(8)))  __bf16 v8bf;
typedef __attribute__((ext_vector_type(8)))  float  v8f;

union F8     { v8f   v; float  e[8];  };
union BF16V  { v16bf v; __bf16 e[16]; };

// WMMA: D = A(16x32 bf16) * B(32x16 bf16) + C(16x16 f32)
__device__ __forceinline__ v8f wmma_bf16(v16bf a, v16bf b, v8f c) {
  return __builtin_amdgcn_wmma_f32_16x16x32_bf16(
      /*neg_a=*/false, a, /*neg_b=*/false, b,
      /*c_mod=*/(short)0, c, /*reuse_a=*/false, /*reuse_b=*/false);
}

// A-matrix 16x32 bf16 fragment in this code's contiguous k order (B frags use the same):
//   lane l (l%16 = row M), K = (l/16)*16 + [0..15] (contiguous)
__device__ __forceinline__ v16bf load_a_frag(const __bf16* rowp, int lhi) {
  v8bf lo = *(const v8bf*)(rowp + lhi * 16);
  v8bf hi = *(const v8bf*)(rowp + lhi * 16 + 8);
  return __builtin_shufflevector(lo, hi, 0,1,2,3,4,5,6,7,8,9,10,11,12,13,14,15);
}

// Async global -> LDS copy, 16B per lane, tracked by ASYNCcnt.
// LDS destination passed as flat pointer into the kernel's __shared__ array:
// low 32 bits are the wave-relative LDS byte offset (LDS aperture, ISA 10.2).
// Passing the pointer into the asm forces the shared array to escape so LDS
// reads staged by this asm are not folded to undef.
__device__ __forceinline__ void async_load_b128(void* lds_ptr, const void* gaddr) {
  unsigned lds_off = (unsigned)(unsigned long long)lds_ptr;
  asm volatile("global_load_async_to_lds_b128 %0, %1, off"
               :: "v"(lds_off), "v"(gaddr), "v"(lds_ptr) : "memory");
}

// ---------------------------------------------------------------------------
// Kernel 1: fp32 -> bf16 conversions (+ weight transposes)
//   Xh/Xg: [4096][1024] row ib = i*2+b  (identical flat order to input h/g)
//   WqT/WkT/WvT: [nd][k]  (transposed so B-fragments are contiguous)
//   WoB: [h][nd]          (already B^T for the output projection)
// ---------------------------------------------------------------------------
__global__ __launch_bounds__(256) void convert_kernel(
    const float* __restrict__ h, const float* __restrict__ g,
    const float* __restrict__ Wq, const float* __restrict__ Wk,
    const float* __restrict__ Wv, const float* __restrict__ Wo,
    __bf16* __restrict__ Xh, __bf16* __restrict__ Xg,
    __bf16* __restrict__ WqT, __bf16* __restrict__ WkT,
    __bf16* __restrict__ WvT, __bf16* __restrict__ WoB)
{
  size_t idx = (size_t)blockIdx.x * blockDim.x + threadIdx.x;
  if (idx < (size_t)MROWS * DMODEL) {
    Xh[idx] = (__bf16)h[idx];
    Xg[idx] = (__bf16)g[idx];
  }
  if (idx < (size_t)DMODEL * DMODEL) {
    int r = (int)(idx >> 10);       // k (d_model dim)
    int c = (int)(idx & 1023);      // nd
    size_t t = (size_t)c * DMODEL + r;
    WqT[t] = (__bf16)Wq[idx];
    WkT[t] = (__bf16)Wk[idx];
    WvT[t] = (__bf16)Wv[idx];
    WoB[idx] = (__bf16)Wo[idx];
  }
}

// ---------------------------------------------------------------------------
// Kernel 2: C[M=4096][N=1024] = A[M][K=1024] * B  (Bt stored [N][K] bf16)
//   WG = 256 thr (8 waves, 4M x 2N). WG tile 64(M) x 128(N), K-step 64.
//   A/B tiles double-buffered in LDS via global_load_async_to_lds_b128.
//   Per K-step: preload all fragments, then issue the 8-WMMA chain.
//   mode 0: bf16 scatter to [b][n][i][d] (+optional bias[nd])  -> q/k/v heads
//   mode 1: f32 row-major [M][N]                               -> attn_out
// ---------------------------------------------------------------------------
#define GEMM_BUF 24576                         // 8KB A + 16KB B per stage
__global__ __launch_bounds__(256) void gemm_bf16(
    const __bf16* __restrict__ A, const __bf16* __restrict__ Bt,
    __bf16* __restrict__ outScatter, const float* __restrict__ bias,
    float* __restrict__ outRow, int mode)
{
  __shared__ __align__(16) char smem[2 * GEMM_BUF];

  const int tid  = threadIdx.x;
  const int lane = tid & 31;
  const int wv   = tid >> 5;
  const int lrw  = lane & 15;
  const int lhi  = lane >> 4;
  const int wm   = wv >> 1;                    // 0..3
  const int wn   = wv & 1;                     // 0..1
  const int m0   = blockIdx.x * 64;
  const int n0   = blockIdx.y * 128;

  auto issue = [&](int k0, int buf) {
    char* base = smem + buf * GEMM_BUF;
    // A tile: 64 rows x 64 K  -> 512 x 16B chunks, 2 per thread
#pragma unroll
    for (int q = 0; q < 2; ++q) {
      int c = tid + q * 256;
      int row = c >> 3, col = c & 7;
      async_load_b128(base + row * 128 + col * 16,
                      A + (size_t)(m0 + row) * DMODEL + k0 + col * 8);
    }
    // B tile: 128 rows x 64 K -> 1024 x 16B chunks, 4 per thread
#pragma unroll
    for (int q = 0; q < 4; ++q) {
      int c = tid + q * 256;
      int row = c >> 3, col = c & 7;
      async_load_b128(base + 8192 + row * 128 + col * 16,
                      Bt + (size_t)(n0 + row) * DMODEL + k0 + col * 8);
    }
  };

  F8 acc[4] = {};
  issue(0, 0);
  const int NT = DMODEL / 64;                  // 16 K-steps
  for (int t = 0; t < NT; ++t) {
    if (t + 1 < NT) {
      issue((t + 1) * 64, (t + 1) & 1);
      asm volatile("s_wait_asynccnt 6" ::: "memory");   // stage t complete
    } else {
      asm volatile("s_wait_asynccnt 0" ::: "memory");
    }
    __syncthreads();
    const __bf16* Abuf = (const __bf16*)(smem + (t & 1) * GEMM_BUF);
    const __bf16* Bbuf = (const __bf16*)(smem + (t & 1) * GEMM_BUF + 8192);
    const __bf16* arow = Abuf + (wm * 16 + lrw) * 64;

    // Preload all fragments for this K-step, then run the WMMA chain so the
    // ds_loads pipeline instead of serializing on s_wait_dscnt per WMMA.
    v16bf af[2];
    af[0] = load_a_frag(arow, lhi);
    af[1] = load_a_frag(arow + 32, lhi);
    v16bf bf[2][4];
#pragma unroll
    for (int kc = 0; kc < 2; ++kc)
#pragma unroll
      for (int s = 0; s < 4; ++s)
        bf[kc][s] = *(const v16bf*)(Bbuf + (wn * 64 + s * 16 + lrw) * 64 + kc * 32 + lhi * 16);
#pragma unroll
    for (int kc = 0; kc < 2; ++kc)
#pragma unroll
      for (int s = 0; s < 4; ++s)
        acc[s].v = wmma_bf16(af[kc], bf[kc][s], acc[s].v);
    __syncthreads();                           // protect buffer before re-issue
  }

  const int wm0 = m0 + wm * 16;
  const int wn0 = n0 + wn * 64;
  if (mode == 0) {
#pragma unroll
    for (int s = 0; s < 4; ++s) {
#pragma unroll
      for (int r = 0; r < 8; ++r) {
        int row = wm0 + lhi * 8 + r;           // ib = i*2+b
        int col = wn0 + s * 16 + lrw;          // nd
        float vv = acc[s].e[r];
        if (bias) vv += bias[col];
        int i = row >> 1, b = row & 1, n = col >> 6, d = col & 63;
        outScatter[((((size_t)b * NHEAD + n) * S_LEN + i) << 6) + d] = (__bf16)vv;
      }
    }
  } else {
#pragma unroll
    for (int s = 0; s < 4; ++s) {
#pragma unroll
      for (int r = 0; r < 8; ++r) {
        int row = wm0 + lhi * 8 + r;
        int col = wn0 + s * 16 + lrw;
        outRow[(size_t)row * DMODEL + col] = acc[s].e[r];
      }
    }
  }
}

// ---------------------------------------------------------------------------
// Kernel 3: flash attention. One wave per 16-row i-tile; all 8 waves of a
//   block share one (b, n) head, so K/V j-tiles (32x64) are staged
//   cooperatively into LDS with async copies, double-buffered.
//   Q/K/V in [b][n][seq][64] bf16. AV out: [ib][nd] bf16 row-major.
// ---------------------------------------------------------------------------
#define KV_BUF 8192                            // 4KB K + 4KB V per stage
__global__ __launch_bounds__(256) void attention_kernel(
    const __bf16* __restrict__ Q, const __bf16* __restrict__ Km,
    const __bf16* __restrict__ Vm, __bf16* __restrict__ AV)
{
  __shared__ __align__(16) char smem[2 * KV_BUF + 8192];   // 16KB KV + 8KB P

  const int tid  = threadIdx.x;
  const int lane = tid & 31;
  const int wv   = tid >> 5;
  const int lrw  = lane & 15;
  const int lhi  = lane >> 4;

  // block -> (b, n, itile base); 16 blocks per head (128 itiles / 8 waves)
  int blk = blockIdx.x;
  const int itb = blk & 15;          blk >>= 4;
  const int n   = blk & (NHEAD - 1); blk >>= 4;
  const int b   = blk;
  const size_t base = ((size_t)b * NHEAD + n) * S_LEN * DHEAD;
  const int i0 = (itb * 8 + wv) * 16;

  __bf16* Pb = (__bf16*)(smem + 2 * KV_BUF) + wv * 512;    // per-wave 16x32 P

  // Q fragments (d 0..31 and 32..63), loaded once per wave
  const __bf16* qrow = Q + base + (size_t)(i0 + lrw) * DHEAD;
  v16bf a0 = load_a_frag(qrow, lhi);
  v16bf a1 = load_a_frag(qrow + 32, lhi);

  auto issue = [&](int jt, int buf) {
    char* bb = smem + buf * KV_BUF;
    int row = tid >> 3, col = tid & 7;         // 32 rows x 8 chunks = 256
    async_load_b128(bb + row * 128 + col * 16,
                    Km + base + (size_t)(jt + row) * DHEAD + col * 8);
    async_load_b128(bb + 4096 + row * 128 + col * 16,
                    Vm + base + (size_t)(jt + row) * DHEAD + col * 8);
  };

  const float scale = 0.125f;                  // 1/sqrt(64)
  float mrow[8], lsum[8], corrs[8];
#pragma unroll
  for (int r = 0; r < 8; ++r) { mrow[r] = -3.0e38f; lsum[r] = 0.0f; }
  F8 o[4] = {};

  issue(0, 0);
  const int NT = S_LEN / 32;                   // 64 j-tiles
  for (int t = 0; t < NT; ++t) {
    if (t + 1 < NT) {
      issue((t + 1) * 32, (t + 1) & 1);
      asm volatile("s_wait_asynccnt 2" ::: "memory");
    } else {
      asm volatile("s_wait_asynccnt 0" ::: "memory");
    }
    __syncthreads();
    const __bf16* Klds = (const __bf16*)(smem + (t & 1) * KV_BUF);
    const __bf16* Vlds = (const __bf16*)(smem + (t & 1) * KV_BUF + 4096);

    // Preload all K fragments, then compute S = Q * K^T for both j blocks
    v16bf kb[2][2];
#pragma unroll
    for (int jb = 0; jb < 2; ++jb) {
      const __bf16* kp = Klds + (jb * 16 + lrw) * DHEAD;
      kb[jb][0] = *(const v16bf*)(kp + lhi * 16);        // d 0..31
      kb[jb][1] = *(const v16bf*)(kp + 32 + lhi * 16);   // d 32..63
    }
    F8 s0 = {}, s1 = {};
    s0.v = wmma_bf16(a0, kb[0][0], s0.v);
    s0.v = wmma_bf16(a1, kb[0][1], s0.v);
    s1.v = wmma_bf16(a0, kb[1][0], s1.v);
    s1.v = wmma_bf16(a1, kb[1][1], s1.v);

    // Online softmax (rows in VGPRs, cols across 16-lane halves)
#pragma unroll
    for (int r = 0; r < 8; ++r) {
      float sa = s0.e[r] * scale;
      float sb = s1.e[r] * scale;
      float mx = fmaxf(sa, sb);
#pragma unroll
      for (int off = 1; off < 16; off <<= 1)
        mx = fmaxf(mx, __shfl_xor(mx, off, 32));
      float mnew = fmaxf(mrow[r], mx);
      float corr = __expf(mrow[r] - mnew);
      mrow[r] = mnew;
      float p0 = __expf(sa - mnew);
      float p1 = __expf(sb - mnew);
      float rs = p0 + p1;
#pragma unroll
      for (int off = 1; off < 16; off <<= 1)
        rs += __shfl_xor(rs, off, 32);
      lsum[r] = lsum[r] * corr + rs;
      corrs[r] = corr;
      Pb[(lhi * 8 + r) * 32 + lrw]      = (__bf16)p0;
      Pb[(lhi * 8 + r) * 32 + 16 + lrw] = (__bf16)p1;
    }
#pragma unroll
    for (int s = 0; s < 4; ++s)
#pragma unroll
      for (int r = 0; r < 8; ++r) o[s].e[r] *= corrs[r];

    // Reshape P (C-layout -> A-layout) via per-wave LDS region
    asm volatile("s_wait_dscnt 0" ::: "memory");
    v8bf plo = *(const v8bf*)(Pb + lrw * 32 + lhi * 16);
    v8bf phi = *(const v8bf*)(Pb + lrw * 32 + lhi * 16 + 8);
    v16bf pa = __builtin_shufflevector(plo, phi,
        0,1,2,3,4,5,6,7,8,9,10,11,12,13,14,15);

    // Preload V fragments (B-frag: lane col d = s*16 + lrw, rows j = lhi*16+e)
    BF16V bv[4];
#pragma unroll
    for (int s = 0; s < 4; ++s) {
      const __bf16* vp = Vlds + (lhi * 16) * DHEAD + s * 16 + lrw;
#pragma unroll
      for (int e = 0; e < 16; ++e) bv[s].e[e] = vp[e * DHEAD];
    }
    // O += P * V
#pragma unroll
    for (int s = 0; s < 4; ++s)
      o[s].v = wmma_bf16(pa, bv[s].v, o[s].v);
    __syncthreads();                           // protect KV buffer re-issue
  }

  // Normalize and store attn_vec row-major [ib][nd] as bf16
  float inv[8];
#pragma unroll
  for (int r = 0; r < 8; ++r) inv[r] = 1.0f / lsum[r];
#pragma unroll
  for (int s = 0; s < 4; ++s) {
#pragma unroll
    for (int r = 0; r < 8; ++r) {
      int i = i0 + lhi * 8 + r;
      int col = n * DHEAD + s * 16 + lrw;
      AV[(size_t)(i * BATCH_N + b) * DMODEL + col] = (__bf16)(o[s].e[r] * inv[r]);
    }
  }
}

// ---------------------------------------------------------------------------
// Kernel 4: y = LayerNorm(attn_out + x) per 1024-wide row
// ---------------------------------------------------------------------------
__device__ __forceinline__ float block_reduce_sum(float v, float* sred) {
  int lane = threadIdx.x & 31, wv = threadIdx.x >> 5;
#pragma unroll
  for (int off = 16; off > 0; off >>= 1) v += __shfl_xor(v, off, 32);
  if (lane == 0) sred[wv] = v;
  __syncthreads();
  float tot = sred[0];
#pragma unroll
  for (int w = 1; w < 8; ++w) tot += sred[w];
  return tot;
}

__global__ __launch_bounds__(256) void ln_kernel(
    const float* __restrict__ AO, const float* __restrict__ X,
    const float* __restrict__ gamma, const float* __restrict__ beta,
    float* __restrict__ out)
{
  __shared__ float sred[8];
  const size_t row = blockIdx.x;
  const float* ao = AO + row * DMODEL;
  const float* x  = X  + row * DMODEL;
  float*       o  = out + row * DMODEL;
  const int tid = threadIdx.x;

  float t[4], sum = 0.0f;
#pragma unroll
  for (int e = 0; e < 4; ++e) {
    int c = tid + e * 256;
    t[e] = ao[c] + x[c];
    sum += t[e];
  }
  float mu = block_reduce_sum(sum, sred) * (1.0f / DMODEL);
  __syncthreads();
  float vs = 0.0f;
#pragma unroll
  for (int e = 0; e < 4; ++e) { float d = t[e] - mu; vs += d * d; }
  float var = block_reduce_sum(vs, sred) * (1.0f / DMODEL);
  float rstd = rsqrtf(var + 1e-5f);
#pragma unroll
  for (int e = 0; e < 4; ++e) {
    int c = tid + e * 256;
    o[c] = gamma[c] * (t[e] - mu) * rstd + beta[c];
  }
}

// ---------------------------------------------------------------------------
// Host launcher
// ---------------------------------------------------------------------------
extern "C" void kernel_launch(void* const* d_in, const int* in_sizes, int n_in,
                              void* d_out, int out_size, void* d_ws, size_t ws_size,
                              hipStream_t stream) {
  const float* h  = (const float*)d_in[0];
  const float* g  = (const float*)d_in[1];
  const float* Wq = (const float*)d_in[2];
  const float* Wk = (const float*)d_in[3];
  const float* Wv = (const float*)d_in[4];
  const float* Wo = (const float*)d_in[5];
  const float* r_w_bias = (const float*)d_in[6];
  const float* ln_gamma = (const float*)d_in[7];
  const float* ln_beta  = (const float*)d_in[8];
  float* out = (float*)d_out;

  // Workspace carve-up (~104 MiB total)
  char* ws = (char*)d_ws;
  const size_t SZ_X  = (size_t)MROWS * DMODEL * sizeof(__bf16);   // 8 MiB
  const size_t SZ_W  = (size_t)DMODEL * DMODEL * sizeof(__bf16);  // 2 MiB
  const size_t SZ_H  = (size_t)BATCH_N * NHEAD * S_LEN * DHEAD * sizeof(__bf16); // 8 MiB
  const size_t SZ_AO = (size_t)MROWS * DMODEL * sizeof(float);    // 16 MiB
  size_t off = 0;
  __bf16* Xh  = (__bf16*)(ws + off); off += SZ_X;
  __bf16* Xg  = (__bf16*)(ws + off); off += SZ_X;
  __bf16* WqT = (__bf16*)(ws + off); off += SZ_W;
  __bf16* WkT = (__bf16*)(ws + off); off += SZ_W;
  __bf16* WvT = (__bf16*)(ws + off); off += SZ_W;
  __bf16* WoB = (__bf16*)(ws + off); off += SZ_W;
  __bf16* Qh  = (__bf16*)(ws + off); off += SZ_H;
  __bf16* Qg  = (__bf16*)(ws + off); off += SZ_H;
  __bf16* Kh  = (__bf16*)(ws + off); off += SZ_H;
  __bf16* Vh  = (__bf16*)(ws + off); off += SZ_H;
  __bf16* AVh = (__bf16*)(ws + off); off += SZ_X;
  __bf16* AVg = (__bf16*)(ws + off); off += SZ_X;
  float*  AOh = (float*)(ws + off);  off += SZ_AO;
  float*  AOg = (float*)(ws + off);  off += SZ_AO;

  // 1) conversions / transposes
  convert_kernel<<<(MROWS * DMODEL + 255) / 256, 256, 0, stream>>>(
      h, g, Wq, Wk, Wv, Wo, Xh, Xg, WqT, WkT, WvT, WoB);

  // 2) projections (async-staged WMMA GEMMs), scatter to [b][n][i][d]
  dim3 ggrid(MROWS / 64, DMODEL / 128);
  gemm_bf16<<<ggrid, 256, 0, stream>>>(Xh, WkT, Kh, nullptr,  nullptr, 0);
  gemm_bf16<<<ggrid, 256, 0, stream>>>(Xh, WvT, Vh, nullptr,  nullptr, 0);
  gemm_bf16<<<ggrid, 256, 0, stream>>>(Xh, WqT, Qh, r_w_bias, nullptr, 0);
  gemm_bf16<<<ggrid, 256, 0, stream>>>(Xg, WqT, Qg, r_w_bias, nullptr, 0);

  // 3) flash attention, both streams share K/V from h
  const int attn_blocks = (BATCH_N * NHEAD * (S_LEN / 16)) / 8;  // 512
  attention_kernel<<<attn_blocks, 256, 0, stream>>>(Qh, Kh, Vh, AVh);
  attention_kernel<<<attn_blocks, 256, 0, stream>>>(Qg, Kh, Vh, AVg);

  // 4) output projection (f32 row-major)
  gemm_bf16<<<ggrid, 256, 0, stream>>>(AVh, WoB, nullptr, nullptr, AOh, 1);
  gemm_bf16<<<ggrid, 256, 0, stream>>>(AVg, WoB, nullptr, nullptr, AOg, 1);

  // 5) residual + layernorm
  ln_kernel<<<MROWS, 256, 0, stream>>>(AOh, h, ln_gamma, ln_beta, out);
  ln_kernel<<<MROWS, 256, 0, stream>>>(AOg, g, ln_gamma, ln_beta,
                                       out + (size_t)MROWS * DMODEL);
}
